// CausalSelfAttention_27857157882038
// MI455X (gfx1250) — matrix operand order repaired and checked by reading, not compiled, not run on hardware
//
#include <hip/hip_runtime.h>

#define D_MODEL 1024
#define N_HEADS 16
#define HEAD_DIM 64
#define SEQ 2048
#define BATCH 4
#define M_TOKENS (BATCH * SEQ) /* 8192 */

typedef __attribute__((ext_vector_type(16))) __bf16 v16bf;
typedef __attribute__((ext_vector_type(8)))  float  v8f;
typedef __attribute__((ext_vector_type(4))) unsigned int u32x4;
typedef __attribute__((ext_vector_type(8))) int          i32x8;
typedef __attribute__((ext_vector_type(4))) int          i32x4;

union FragBF { v16bf v; unsigned int u[8]; };
union AccF  { v8f   v; float f[8]; };

__device__ __forceinline__ unsigned short f2bf(float x) {
  unsigned int b = __float_as_uint(x);
  b += 0x7FFFu + ((b >> 16) & 1u);   // round-to-nearest-even
  return (unsigned short)(b >> 16);
}

// ---------------------------------------------------------------------------
// Tensor Data Mover: 2D tile (bf16, data_size=2B) global -> LDS.
// Descriptor layout per CDNA5 ISA ch.8 (D# groups 0/1; groups 2/3 zero = 2D).
// ---------------------------------------------------------------------------
__device__ __forceinline__ void tdm_load_2d_bf16(
    unsigned int lds_off, const void* gptr, unsigned int tile_d0,
    unsigned int tile_d1, unsigned int tensor_d0, unsigned int tensor_d1,
    unsigned int stride_d0) {
  const unsigned long long ga = (unsigned long long)(uintptr_t)gptr;
  u32x4 g0;
  g0[0] = 1u;                                   // count=1, user descriptor
  g0[1] = lds_off;                              // LDS byte address
  g0[2] = (unsigned int)(ga & 0xFFFFFFFFu);     // global_addr[31:0]
  g0[3] = (unsigned int)((ga >> 32) & 0x1FFFFFFu) | (2u << 30);  // type=2
  i32x8 g1;
  g1[0] = (int)(1u << 16);                      // wg_mask=0, data_size=1 (2B)
  g1[1] = (int)(tensor_d0 << 16);               // tensor_dim0[15:0] @ [63:48]
  g1[2] = (int)((tensor_d0 >> 16) | (tensor_d1 << 16));
  g1[3] = (int)((tensor_d1 >> 16) | (tile_d0 << 16));   // tile_dim0 @[127:112]
  g1[4] = (int)(tile_d1 & 0xFFFFu);             // tile_dim1; tile_dim2=0
  g1[5] = (int)stride_d0;                       // tensor_dim0_stride[31:0]
  g1[6] = 0;                                    // stride hi, dim1_stride lo
  g1[7] = 0;
  i32x4 z4 = {0, 0, 0, 0};
#if defined(__clang_major__) && (__clang_major__ >= 23)
  i32x8 z8 = {0, 0, 0, 0, 0, 0, 0, 0};
  __builtin_amdgcn_tensor_load_to_lds(g0, g1, z4, z4, z8, 0);
#else
  __builtin_amdgcn_tensor_load_to_lds(g0, g1, z4, z4, 0);
#endif
}

// A-operand gather, 16-bit A 16x32 (ISA 7.12.2):
// lanes 0-15: VGPR0-3 -> K=0..7, VGPR4-7 -> K=16..23; lanes 16-31: +8
__device__ __forceinline__ void fragA(const unsigned short* base, int row,
                                      int stride, int k0, FragBF& f) {
  const int lane = threadIdx.x & 31;
  const int hi = lane >> 4;
  const unsigned short* p = base + row * stride + k0 + hi * 8;
#pragma unroll
  for (int v = 0; v < 8; ++v) {
    const int kk = (v < 4) ? (2 * v) : (8 + 2 * v);
    f.u[v] = *(const unsigned int*)(p + kk);
  }
}

// B-operand gather, 16-bit B 32x16: lanes 0-15 hold K=0..7,16..23; lanes 16-31: +8.
// base is [n][k] row-major; caller passes row = n (incl. lane%16).
__device__ __forceinline__ void fragB(const unsigned short* base, int row,
                                      int stride, int k0, FragBF& f) {
  const int lane = threadIdx.x & 31;
  const int hi = lane >> 4;
  const unsigned short* p = base + row * stride + k0 + hi * 8;
#pragma unroll
  for (int v = 0; v < 8; ++v) {
    const int kk = (v < 4) ? (2 * v) : (8 + 2 * v);
    f.u[v] = *(const unsigned int*)(p + kk);
  }
}

// ---------------------------------------------------------------------------
// Kernel 1: qkv = x @ W_qkv + b ; scatter Q,K,V (bf16) into [B,H,T,64]
// Block tile 128x128, K-chunks of 32, 8 waves, wave tile 32x64.
// (fp32 -> bf16 conversion happens while staging, so no TDM here)
// ---------------------------------------------------------------------------
__global__ __launch_bounds__(256) void qkv_wmma_kernel(
    const float* __restrict__ X, const float* __restrict__ W,
    const float* __restrict__ bias, unsigned short* __restrict__ Qb,
    unsigned short* __restrict__ Kb, unsigned short* __restrict__ Vb) {
  __shared__ __align__(16) unsigned short lA[128 * 32];
  __shared__ __align__(16) unsigned short lB[128 * 32];
  const int tid = threadIdx.x;
  const int lane = tid & 31, wid = tid >> 5;
  const int ln = lane & 15, hi = lane >> 4;
  const int m0 = blockIdx.y * 128;
  const int n0 = blockIdx.x * 128;
  const int wm = (wid & 3) * 32;
  const int wn = (wid >> 2) * 64;

  AccF acc[2][4] = {};

  for (int kb = 0; kb < D_MODEL / 32; ++kb) {
    // stage A: 128x32 fp32 -> bf16 [m][k]
#pragma unroll
    for (int p = 0; p < 4; ++p) {
      const int idx = p * 256 + tid;
      const int row = idx >> 3, c4 = (idx & 7) * 4;
      const float4 f =
          *(const float4*)(X + (size_t)(m0 + row) * D_MODEL + kb * 32 + c4);
      unsigned short* d = lA + row * 32 + c4;
      d[0] = f2bf(f.x); d[1] = f2bf(f.y); d[2] = f2bf(f.z); d[3] = f2bf(f.w);
    }
    // stage B: 32x128 fp32 -> bf16, transposed to [n][k]
#pragma unroll
    for (int p = 0; p < 4; ++p) {
      const int idx = p * 256 + tid;
      const int k = idx >> 5, nf = (idx & 31) * 4;
      const float4 f =
          *(const float4*)(W + (size_t)(kb * 32 + k) * 3072 + n0 + nf);
      lB[(nf + 0) * 32 + k] = f2bf(f.x);
      lB[(nf + 1) * 32 + k] = f2bf(f.y);
      lB[(nf + 2) * 32 + k] = f2bf(f.z);
      lB[(nf + 3) * 32 + k] = f2bf(f.w);
    }
    if (kb + 1 < D_MODEL / 32) {  // -> global_prefetch_b8
      __builtin_prefetch(X + (size_t)(m0 + (tid >> 3)) * D_MODEL + kb * 32 + 32, 0, 0);
      __builtin_prefetch(W + (size_t)(kb * 32 + 32 + (tid >> 5)) * 3072 + n0, 0, 0);
    }
    __syncthreads();

    FragBF fa[2], fb[4];
#pragma unroll
    for (int mt = 0; mt < 2; ++mt) fragA(lA, wm + mt * 16 + ln, 32, 0, fa[mt]);
#pragma unroll
    for (int nt = 0; nt < 4; ++nt) fragB(lB, wn + nt * 16 + ln, 32, 0, fb[nt]);
#pragma unroll
    for (int mt = 0; mt < 2; ++mt)
#pragma unroll
      for (int nt = 0; nt < 4; ++nt)
        acc[mt][nt].v = __builtin_amdgcn_wmma_f32_16x16x32_bf16(
            false, fa[mt].v, false, fb[nt].v, (short)0, acc[mt][nt].v, false,
            false);
    __syncthreads();
  }

  // epilogue: +bias, split into Q/K/V head-major bf16
#pragma unroll
  for (int mt = 0; mt < 2; ++mt)
#pragma unroll
    for (int nt = 0; nt < 4; ++nt)
#pragma unroll
      for (int r = 0; r < 8; ++r) {
        const int gm = m0 + wm + mt * 16 + r + hi * 8;   // token index b*T+t
        const int gn = n0 + wn + nt * 16 + ln;           // col in [0,3072)
        const float val = acc[mt][nt].f[r] + bias[gn];
        const int which = gn >> 10;
        const int h = (gn >> 6) & 15, d = gn & 63;
        const int b = gm >> 11, t = gm & (SEQ - 1);
        unsigned short* dst = (which == 0) ? Qb : ((which == 1) ? Kb : Vb);
        dst[(((size_t)(b * N_HEADS + h)) * SEQ + t) * HEAD_DIM + d] = f2bf(val);
      }
}

// ---------------------------------------------------------------------------
// Kernel 2: causal flash attention. 1 wave per (b,h, 16-query tile).
// K block staged by the Tensor Data Mover; key blocks of 32.
// ---------------------------------------------------------------------------
__global__ __launch_bounds__(32) void attn_wmma_kernel(
    const unsigned short* __restrict__ Qb, const unsigned short* __restrict__ Kb,
    const unsigned short* __restrict__ Vb, unsigned short* __restrict__ Yb) {
  __shared__ __align__(16) unsigned short lK[32 * 64];   // [key][d]
  __shared__ __align__(16) unsigned short lVt[64 * 32];  // [d][key]
  __shared__ __align__(16) unsigned short lP[16 * 32];   // [q][key]
  const int lane = threadIdx.x & 31;
  const int ln = lane & 15, hi = lane >> 4;
  const int q0 = blockIdx.x * 16;
  const int bh = blockIdx.y;
  const unsigned int lKoff = (unsigned int)(uintptr_t)(&lK[0]);

  // Q A-fragments straight from global (bf16, [bh][t][d])
  FragBF fq[2];
#pragma unroll
  for (int c = 0; c < 2; ++c)
    fragA(Qb + ((size_t)bh * SEQ + q0 + ln) * HEAD_DIM, 0, HEAD_DIM, c * 32,
          fq[c]);

  float m[8], l[8];
  AccF acc[4] = {};
#pragma unroll
  for (int r = 0; r < 8; ++r) { m[r] = -1e30f; l[r] = 0.0f; }

  const int nblocks = ((q0 + 15) >> 5) + 1;  // causal: only needed key blocks
  for (int kb = 0; kb < nblocks; ++kb) {
    const int k0 = kb * 32;
    // stage K block [key][d] via Tensor Data Mover (32 rows x 64 bf16)
    tdm_load_2d_bf16(lKoff, Kb + ((size_t)bh * SEQ + k0) * HEAD_DIM,
                     /*tile_d0=*/HEAD_DIM, /*tile_d1=*/32,
                     /*tensor_d0=*/HEAD_DIM, /*tensor_d1=*/SEQ,
                     /*stride_d0=*/HEAD_DIM);
    // stage V block transposed [d][key] (transpose: manual)
#pragma unroll 4
    for (int i = 0; i < 64; ++i) {
      const int idx = i * 32 + lane;  // 2048 ushorts
      const int key = idx >> 6, d = idx & 63;
      lVt[d * 32 + key] = Vb[((size_t)bh * SEQ + k0 + key) * HEAD_DIM + d];
    }
    __builtin_amdgcn_s_wait_tensorcnt(0);
    __syncthreads();

    // S = Q K^T : two 16x16 N-tiles, Dh=64 -> two k-chunks each
    AccF s[2] = {};
#pragma unroll
    for (int nt = 0; nt < 2; ++nt)
#pragma unroll
      for (int c = 0; c < 2; ++c) {
        FragBF fk;
        fragB(lK, nt * 16 + ln, 64, c * 32, fk);
        s[nt].v = __builtin_amdgcn_wmma_f32_16x16x32_bf16(
            false, fq[c].v, false, fk.v, (short)0, s[nt].v, false, false);
      }

    // online softmax (row = r + 8*hi lives across the 16 lanes of one half)
#pragma unroll
    for (int r = 0; r < 8; ++r) {
      const int row = q0 + r + hi * 8;
      float s0 = s[0].f[r] * 0.125f;  // 1/sqrt(64)
      float s1 = s[1].f[r] * 0.125f;
      if (k0 + ln > row) s0 = -1e30f;
      if (k0 + 16 + ln > row) s1 = -1e30f;
      float mx = fmaxf(s0, s1);
#pragma unroll
      for (int off = 1; off < 16; off <<= 1)
        mx = fmaxf(mx, __shfl_xor(mx, off, 32));
      const float mn = fmaxf(m[r], mx);
      const float p0 = __expf(s0 - mn);
      const float p1 = __expf(s1 - mn);
      float rs = p0 + p1;
#pragma unroll
      for (int off = 1; off < 16; off <<= 1) rs += __shfl_xor(rs, off, 32);
      const float alpha = __expf(m[r] - mn);
      l[r] = l[r] * alpha + rs;
      m[r] = mn;
#pragma unroll
      for (int nt = 0; nt < 4; ++nt) acc[nt].f[r] *= alpha;
      lP[(r + hi * 8) * 32 + ln] = f2bf(p0);       // re-layout P via LDS
      lP[(r + hi * 8) * 32 + 16 + ln] = f2bf(p1);
    }
    __syncthreads();

    // acc += P V
    FragBF fp;
    fragA(lP, ln, 32, 0, fp);
#pragma unroll
    for (int nt = 0; nt < 4; ++nt) {
      FragBF fv;
      fragB(lVt, nt * 16 + ln, 32, 0, fv);
      acc[nt].v = __builtin_amdgcn_wmma_f32_16x16x32_bf16(
          false, fp.v, false, fv.v, (short)0, acc[nt].v, false, false);
    }
    __syncthreads();
  }

  // normalize and write y back token-major [B,T,C] bf16
  const int b = bh >> 4, h = bh & 15;
#pragma unroll
  for (int nt = 0; nt < 4; ++nt)
#pragma unroll
    for (int r = 0; r < 8; ++r) {
      const int t = q0 + r + hi * 8;
      const float v = acc[nt].f[r] / l[r];
      Yb[(((size_t)b * SEQ + t) * N_HEADS + h) * HEAD_DIM + nt * 16 + ln] =
          f2bf(v);
    }
}

// ---------------------------------------------------------------------------
// Kernel 3: out = y @ W_proj + b_proj (fp32 out). A tile staged by TDM.
// ---------------------------------------------------------------------------
__global__ __launch_bounds__(256) void proj_wmma_kernel(
    const unsigned short* __restrict__ Yb, const float* __restrict__ W,
    const float* __restrict__ bias, float* __restrict__ Out) {
  __shared__ __align__(16) unsigned short lA[128 * 32];
  __shared__ __align__(16) unsigned short lB[128 * 32];
  const int tid = threadIdx.x;
  const int lane = tid & 31, wid = tid >> 5;
  const int ln = lane & 15, hi = lane >> 4;
  const int m0 = blockIdx.y * 128;
  const int n0 = blockIdx.x * 128;
  const int wm = (wid & 3) * 32;
  const int wn = (wid >> 2) * 64;
  const unsigned int lAoff = (unsigned int)(uintptr_t)(&lA[0]);

  AccF acc[2][4] = {};

  for (int kb = 0; kb < D_MODEL / 32; ++kb) {
    // stage A (already bf16): 128x32 tile via Tensor Data Mover, wave 0 only
    if (wid == 0) {
      tdm_load_2d_bf16(lAoff, Yb + (size_t)m0 * D_MODEL + kb * 32,
                       /*tile_d0=*/32, /*tile_d1=*/128,
                       /*tensor_d0=*/D_MODEL, /*tensor_d1=*/M_TOKENS,
                       /*stride_d0=*/D_MODEL);
    }
    // stage B: fp32 -> bf16 transposed [n][k]
#pragma unroll
    for (int p = 0; p < 4; ++p) {
      const int idx = p * 256 + tid;
      const int k = idx >> 5, nf = (idx & 31) * 4;
      const float4 f =
          *(const float4*)(W + (size_t)(kb * 32 + k) * D_MODEL + n0 + nf);
      lB[(nf + 0) * 32 + k] = f2bf(f.x);
      lB[(nf + 1) * 32 + k] = f2bf(f.y);
      lB[(nf + 2) * 32 + k] = f2bf(f.z);
      lB[(nf + 3) * 32 + k] = f2bf(f.w);
    }
    __builtin_amdgcn_s_wait_tensorcnt(0);
    __syncthreads();

    FragBF fa[2], fb[4];
#pragma unroll
    for (int mt = 0; mt < 2; ++mt) fragA(lA, wm + mt * 16 + ln, 32, 0, fa[mt]);
#pragma unroll
    for (int nt = 0; nt < 4; ++nt) fragB(lB, wn + nt * 16 + ln, 32, 0, fb[nt]);
#pragma unroll
    for (int mt = 0; mt < 2; ++mt)
#pragma unroll
      for (int nt = 0; nt < 4; ++nt)
        acc[mt][nt].v = __builtin_amdgcn_wmma_f32_16x16x32_bf16(
            false, fa[mt].v, false, fb[nt].v, (short)0, acc[mt][nt].v, false,
            false);
    __syncthreads();
  }

#pragma unroll
  for (int mt = 0; mt < 2; ++mt)
#pragma unroll
    for (int nt = 0; nt < 4; ++nt)
#pragma unroll
      for (int r = 0; r < 8; ++r) {
        const int gm = m0 + wm + mt * 16 + r + hi * 8;
        const int gn = n0 + wn + nt * 16 + ln;
        Out[(size_t)gm * D_MODEL + gn] = acc[mt][nt].f[r] + bias[gn];
      }
}

// ---------------------------------------------------------------------------
extern "C" void kernel_launch(void* const* d_in, const int* in_sizes, int n_in,
                              void* d_out, int out_size, void* d_ws,
                              size_t ws_size, hipStream_t stream) {
  const float* x = (const float*)d_in[0];
  const float* W_qkv = (const float*)d_in[1];
  const float* b_qkv = (const float*)d_in[2];
  const float* W_proj = (const float*)d_in[3];
  const float* b_proj = (const float*)d_in[4];
  float* out = (float*)d_out;

  const size_t per = (size_t)BATCH * N_HEADS * SEQ * HEAD_DIM;  // 8388608
  unsigned short* Qb = (unsigned short*)d_ws;
  unsigned short* Kb = Qb + per;
  unsigned short* Vb = Kb + per;
  unsigned short* Yb = Vb + per;

  qkv_wmma_kernel<<<dim3(3072 / 128, M_TOKENS / 128), 256, 0, stream>>>(
      x, W_qkv, b_qkv, Qb, Kb, Vb);
  attn_wmma_kernel<<<dim3(SEQ / 16, BATCH * N_HEADS), 32, 0, stream>>>(
      Qb, Kb, Vb, Yb);
  proj_wmma_kernel<<<dim3(D_MODEL / 128, M_TOKENS / 128), 256, 0, stream>>>(
      Yb, W_proj, b_proj, out);
}
